// MoELayer_25795573580236
// MI455X (gfx1250) — matrix operand order repaired and checked by reading, not compiled, not run on hardware
//
#include <hip/hip_runtime.h>
#include <math.h>

// ---------------------------------------------------------------------------
// MoE layer for gfx1250 (MI455X). Top-2 routed expert FFN with bf16 WMMA.
// Workspace layout (requires ws_size >= ~143 MB):
//   [0,   64MB)   W1^T  bf16  [E][4096][1024]
//   [64,  128MB)  W2^T  bf16  [E][1024][4096]
//   [128, 136MB)  x     bf16  [4096][1024]
//   [136MB, +136KB) routing metadata (Meta struct)
// ---------------------------------------------------------------------------

typedef __attribute__((ext_vector_type(16))) __bf16 v16bf;
typedef __attribute__((ext_vector_type(8)))  float  v8f;
typedef unsigned short u16;
typedef unsigned int   u32;

constexpr int D        = 1024;
constexpr int FF       = 4096;
constexpr int NE       = 8;
constexpr int NTOK     = 4096;          // B*S
constexpr int SLOTS    = NTOK * 2;      // top-2
constexpr int MT       = 32;            // token rows per GEMM tile
constexpr int MAXTILES = SLOTS / MT + NE;   // 264 worst case
constexpr int SA       = 1032;          // LDS x-tile stride (halves), 16B aligned
constexpr int SH       = 72;            // LDS h-tile stride (halves), 16B aligned

struct Meta {
  int   topIdx[SLOTS];
  float topW[SLOTS];
  int   counts[NE];
  int   cursors[NE];
  int   fcount[NE];
  float P[NE];
  int   totalTiles;
  int   pad[3];
  int   entries[MAXTILES * MT];
  float entryW[MAXTILES * MT];
  int   tileExpert[MAXTILES];
  int   tileRow[MAXTILES];
  int   tileValid[MAXTILES];
};

constexpr size_t OFF_W1T  = 0;
constexpr size_t OFF_W2T  = 67108864ull;                 // 8*4096*1024*2
constexpr size_t OFF_XB   = 134217728ull;
constexpr size_t OFF_META = 142606336ull;                // + 4096*1024*2

static __device__ __forceinline__ u16 f2bf(float f) {
  u32 u = __float_as_uint(f);
  u32 r = u + 0x7fffu + ((u >> 16) & 1u);     // round-to-nearest-even
  return (u16)(r >> 16);
}

// A fragment (16x32, MxK): lane<16 holds K {k0..k0+15}, lane>=16 holds K {k0+16..k0+31} of row (lane&15)
static __device__ __forceinline__ v16bf load_fragA(const u16* row, int k0, int sk8) {
  union { uint4 u[2]; v16bf v; } f;
  f.u[0] = *(const uint4*)(row + k0 + 2 * sk8);
  f.u[1] = *(const uint4*)(row + k0 + 2 * sk8 + 8);
  return f.v;
}

// B fragment (32x16, KxN): lane<16 holds K {k0..k0+15}, lane>=16 holds K {k0+16..k0+31}
// of column (lane&15).  "row" points at B^T row = column n (K-contiguous).
static __device__ __forceinline__ v16bf load_fragB(const u16* row, int k0, int sk16) {
  union { uint4 u[2]; v16bf v; } f;
  f.u[0] = *(const uint4*)(row + k0 + sk16);
  f.u[1] = *(const uint4*)(row + k0 + sk16 + 8);
  return f.v;
}

static __device__ __forceinline__ v8f wmma_bf16(v16bf a, v16bf b, v8f c) {
  return __builtin_amdgcn_wmma_f32_16x16x32_bf16(false, a, false, b, (short)0, c,
                                                 false, false);
}

// --------------------------- helper kernels --------------------------------

__global__ void k_zero_f32(float* p, int n) {
  int i = blockIdx.x * 256 + threadIdx.x;
  if (i < n) p[i] = 0.f;
}

__global__ void k_zero_meta(Meta* meta) {
  int i = threadIdx.x;
  if (i < 36) ((int*)meta->counts)[i] = 0;   // counts, cursors, fcount, P, totalTiles, pad
}

__global__ void k_xcast(const float* __restrict__ x, u16* __restrict__ xb, int n) {
  int i = blockIdx.x * 256 + threadIdx.x;
  if (i < n) xb[i] = f2bf(x[i]);
}

// [R][C] f32  ->  [C][R] bf16, per expert (blockIdx.z)
__global__ void k_transpose_bf16(const float* __restrict__ src, u16* __restrict__ dst,
                                 int R, int C) {
  __shared__ float tile[32][33];
  const float* s = src + (size_t)blockIdx.z * R * C;
  u16*         d = dst + (size_t)blockIdx.z * R * C;
  int c0 = blockIdx.x * 32, r0 = blockIdx.y * 32;
  int tx = threadIdx.x & 31, ty = threadIdx.x >> 5;
#pragma unroll
  for (int i = 0; i < 4; i++)
    tile[ty + 8 * i][tx] = s[(size_t)(r0 + ty + 8 * i) * C + c0 + tx];
  __syncthreads();
#pragma unroll
  for (int i = 0; i < 4; i++)
    d[(size_t)(c0 + ty + 8 * i) * R + r0 + tx] = f2bf(tile[tx][ty + 8 * i]);
}

// Router: one wave32 per token. softmax over 8 experts, top-2, counts, aux stats.
__global__ void k_router(const float* __restrict__ x, const float* __restrict__ Wr,
                         const float* __restrict__ br, Meta* meta) {
  int lane = threadIdx.x & 31;
  int wid  = threadIdx.x >> 5;
  int tok  = blockIdx.x * 8 + wid;
  if (tok >= NTOK) return;
  float a[NE];
#pragma unroll
  for (int j = 0; j < NE; j++) a[j] = 0.f;
  const float* xr = x + (size_t)tok * D;
  for (int d = lane; d < D; d += 32) {
    float        xv = xr[d];
    const float4* w4 = (const float4*)(Wr + d * NE);
    float4 wa = w4[0], wb = w4[1];
    a[0] += xv * wa.x; a[1] += xv * wa.y; a[2] += xv * wa.z; a[3] += xv * wa.w;
    a[4] += xv * wb.x; a[5] += xv * wb.y; a[6] += xv * wb.z; a[7] += xv * wb.w;
  }
#pragma unroll
  for (int j = 0; j < NE; j++)
    for (int off = 16; off > 0; off >>= 1) a[j] += __shfl_xor(a[j], off, 32);
  if (lane == 0) {
    float p[NE];
    float m = -1e30f;
#pragma unroll
    for (int j = 0; j < NE; j++) { p[j] = a[j] + br[j]; m = fmaxf(m, p[j]); }
    float s = 0.f;
#pragma unroll
    for (int j = 0; j < NE; j++) { p[j] = expf(p[j] - m); s += p[j]; }
    float inv = 1.f / s;
#pragma unroll
    for (int j = 0; j < NE; j++) p[j] *= inv;
    int i0 = 0; float v0 = p[0];
#pragma unroll
    for (int j = 1; j < NE; j++) if (p[j] > v0) { v0 = p[j]; i0 = j; }
    int i1 = -1; float v1 = -1e30f;
#pragma unroll
    for (int j = 0; j < NE; j++) if (j != i0 && p[j] > v1) { v1 = p[j]; i1 = j; }
    float den = fmaxf(v0 + v1, 1e-9f);
    meta->topIdx[tok * 2 + 0] = i0;
    meta->topIdx[tok * 2 + 1] = i1;
    meta->topW[tok * 2 + 0]   = v0 / den;
    meta->topW[tok * 2 + 1]   = v1 / den;
    atomicAdd(&meta->counts[i0], 1);
    atomicAdd(&meta->counts[i1], 1);
    atomicAdd(&meta->fcount[i0], 1);
#pragma unroll
    for (int j = 0; j < NE; j++) atomicAdd(&meta->P[j], p[j]);
  }
}

// Single thread: prefix sums, GEMM-tile table, aux loss.
__global__ void k_setup(Meta* meta, float* out) {
  int off = 0, tt = 0;
  for (int e = 0; e < NE; e++) {
    int c = meta->counts[e];
    meta->cursors[e] = off;
    for (int r = 0; r < c; r += MT) {
      meta->tileExpert[tt] = e;
      meta->tileRow[tt]    = off + r;
      meta->tileValid[tt]  = (c - r < MT) ? (c - r) : MT;
      tt++;
    }
    off += c;
  }
  meta->totalTiles = tt;
  float aux = 0.f;
  for (int j = 0; j < NE; j++)
    aux += ((float)meta->fcount[j] / (float)NTOK) * (meta->P[j] / (float)NTOK);
  out[NTOK * D] = (float)NE * aux;
}

__global__ void k_scatter(Meta* meta) {
  int t = blockIdx.x * 256 + threadIdx.x;
  if (t >= NTOK) return;
#pragma unroll
  for (int k = 0; k < 2; k++) {
    int e   = meta->topIdx[t * 2 + k];
    int pos = atomicAdd(&meta->cursors[e], 1);
    meta->entries[pos] = t;
    meta->entryW[pos]  = meta->topW[t * 2 + k];
  }
}

// ------------------------ fused expert FFN (WMMA) ---------------------------
// One block = 32 tokens of one expert, full 1024-wide output.
// 8 waves: wave w owns output columns [w*128, w*128+128).
// Loop over D_FF in 64-wide chunks: h = gelu(x@W1 + b1) -> out += h@W2.
// GEMM1 uses a 2-stage rotating fragment file, GEMM2 a 4-stage rotating
// B-fragment file (distinct named values, compile-time indices, no copies) so
// the allocator cannot collapse them into one buffer and serialize on
// s_wait_loadcnt 0x0.
__global__ void __launch_bounds__(256, 1)
k_moe_ffn(const u16* __restrict__ w1t, const u16* __restrict__ w2t,
          const u16* __restrict__ xb, const float* __restrict__ b1,
          const float* __restrict__ b2, const Meta* __restrict__ meta,
          float* __restrict__ out) {
  extern __shared__ u16 smem[];
  u16*   xA   = smem;                       // [32][SA] bf16
  u16*   hT   = smem + 32 * SA;             // [32][SH] bf16
  int*   sTok = (int*)(smem + 32 * SA + 32 * SH);
  float* sW   = (float*)(sTok + MT);

  int t = blockIdx.x;
  if (t >= meta->totalTiles) return;
  int e     = meta->tileExpert[t];
  int row0  = meta->tileRow[t];
  int valid = meta->tileValid[t];

  int tid = threadIdx.x, lane = tid & 31, wid = tid >> 5;
  if (tid < MT) {
    int tokv; float wv;
    if (tid < valid) { tokv = meta->entries[row0 + tid]; wv = meta->entryW[row0 + tid]; }
    else             { tokv = meta->entries[row0];       wv = 0.f; }
    sTok[tid] = tokv;
    sW[tid]   = wv;
  }
  __syncthreads();
  // stage 32 gathered token rows (bf16) into LDS
  for (int h = tid; h < MT * (D / 8); h += 256) {
    int r = h >> 7, g = h & 127;
    *(uint4*)(xA + r * SA + g * 8) = *(const uint4*)(xb + (size_t)sTok[r] * D + g * 8);
  }
  __syncthreads();

  const u16* w1e = w1t + (size_t)e * FF * D;   // [ff][k]
  const u16* w2e = w2t + (size_t)e * D * FF;   // [d][ff]

  int mi = wid & 1, ni = wid >> 1;             // GEMM1 subtile of this wave
  int l15  = lane & 15;
  int sk8  = (lane < 16) ? 0 : 8;              // A-fragment lane shift
  int sk16 = (lane < 16) ? 0 : 16;             // B-fragment lane shift

  v8f acc[2][8];
#pragma unroll
  for (int i = 0; i < 2; i++)
#pragma unroll
    for (int j = 0; j < 8; j++) acc[i][j] = v8f{0.f, 0.f, 0.f, 0.f, 0.f, 0.f, 0.f, 0.f};

  const u16* aRow  = xA + (mi * 16 + l15) * SA;
  const u16* hRow0 = hT + l15 * SH;
  const u16* hRow1 = hT + (16 + l15) * SH;
  int        hcol  = ni * 16 + l15;
  int        rbase = mi * 16 + ((lane < 16) ? 0 : 8);

  for (int c = 0; c < FF / 64; c++) {
    // ---- GEMM1: h[32x64] = x[32x1024] @ W1[1024x64], wave computes 16x16 subtile
    v8f hacc = v8f{0.f, 0.f, 0.f, 0.f, 0.f, 0.f, 0.f, 0.f};
    const u16* bRow = w1e + (size_t)(c * 64 + ni * 16 + l15) * D;
    // 2-stage rotating pipeline over K (fully unrolled; guards compile-time)
    v16bf fa0 = load_fragA(aRow, 0, sk8);
    v16bf fb0 = load_fragB(bRow, 0, sk16);
    v16bf fa1 = load_fragA(aRow, 32, sk8);
    v16bf fb1 = load_fragB(bRow, 32, sk16);
#pragma unroll
    for (int kb = 0; kb < D; kb += 64) {
      hacc = wmma_bf16(fa0, fb0, hacc);
      if (kb + 64 < D) {
        fa0 = load_fragA(aRow, kb + 64, sk8);
        fb0 = load_fragB(bRow, kb + 64, sk16);
      }
      hacc = wmma_bf16(fa1, fb1, hacc);
      if (kb + 96 < D) {
        fa1 = load_fragA(aRow, kb + 96, sk8);
        fb1 = load_fragB(bRow, kb + 96, sk16);
      }
    }

    // bias + exact GELU -> bf16 -> LDS
    float b1v = b1[e * FF + c * 64 + hcol];
#pragma unroll
    for (int j = 0; j < 8; j++) {
      float v = hacc[j] + b1v;
      float g = 0.5f * v * (1.f + erff(v * 0.70710678118654752f));
      hT[(rbase + j) * SH + hcol] = f2bf(g);
    }
    __syncthreads();

    // ---- GEMM2: out[32 x 128(wave)] += h[32x64] @ W2[64x1024]
#pragma unroll
    for (int kb2 = 0; kb2 < 64; kb2 += 32) {
      v16bf a0 = load_fragA(hRow0, kb2, sk8);
      v16bf a1 = load_fragA(hRow1, kb2, sk8);
      const u16* bbase = w2e + (size_t)(wid * 128 + l15) * FF + c * 64 + kb2;
      // 4-stage rotating B-fragment file over the 8 output subtiles
      v16bf fb[4];
      fb[0] = load_fragB(bbase + (size_t)0 * 16 * FF, 0, sk16);
      fb[1] = load_fragB(bbase + (size_t)1 * 16 * FF, 0, sk16);
      fb[2] = load_fragB(bbase + (size_t)2 * 16 * FF, 0, sk16);
      fb[3] = load_fragB(bbase + (size_t)3 * 16 * FF, 0, sk16);
#pragma unroll
      for (int nt = 0; nt < 8; nt++) {
        acc[0][nt] = wmma_bf16(a0, fb[nt & 3], acc[0][nt]);
        acc[1][nt] = wmma_bf16(a1, fb[nt & 3], acc[1][nt]);
        if (nt + 4 < 8)
          fb[nt & 3] = load_fragB(bbase + (size_t)(nt + 4) * 16 * FF, 0, sk16);
      }
    }
    __syncthreads();
  }

  // epilogue: out[token] += gateWeight * (acc + b2)
  int rhalf = (lane < 16) ? 0 : 8;
#pragma unroll
  for (int nt = 0; nt < 8; nt++) {
    int   dcol = wid * 128 + nt * 16 + l15;
    float b2v  = b2[e * D + dcol];
#pragma unroll
    for (int m2 = 0; m2 < 2; m2++) {
#pragma unroll
      for (int j = 0; j < 8; j++) {
        int r = m2 * 16 + rhalf + j;
        if (r < valid) {
          float val = (acc[m2][nt][j] + b2v) * sW[r];
          atomicAdd(out + (size_t)sTok[r] * D + dcol, val);
        }
      }
    }
  }
}

// ------------------------------ launcher ------------------------------------

extern "C" void kernel_launch(void* const* d_in, const int* in_sizes, int n_in,
                              void* d_out, int out_size, void* d_ws, size_t ws_size,
                              hipStream_t stream) {
  const float* x  = (const float*)d_in[0];
  const float* Wr = (const float*)d_in[1];
  const float* br = (const float*)d_in[2];
  const float* W1 = (const float*)d_in[3];
  const float* b1 = (const float*)d_in[4];
  const float* W2 = (const float*)d_in[5];
  const float* b2 = (const float*)d_in[6];
  float* out = (float*)d_out;
  char*  ws  = (char*)d_ws;

  u16*  w1t  = (u16*)(ws + OFF_W1T);
  u16*  w2t  = (u16*)(ws + OFF_W2T);
  u16*  xbuf = (u16*)(ws + OFF_XB);
  Meta* meta = (Meta*)(ws + OFF_META);

  k_zero_meta<<<1, 64, 0, stream>>>(meta);
  int totOut = NTOK * D + 1;
  k_zero_f32<<<(totOut + 255) / 256, 256, 0, stream>>>(out, totOut);
  k_xcast<<<(NTOK * D) / 256, 256, 0, stream>>>(x, xbuf, NTOK * D);
  // W1 [E][D][FF] -> W1^T [E][FF][D]
  k_transpose_bf16<<<dim3(FF / 32, D / 32, NE), 256, 0, stream>>>(W1, w1t, D, FF);
  // W2 [E][FF][D] -> W2^T [E][D][FF]
  k_transpose_bf16<<<dim3(D / 32, FF / 32, NE), 256, 0, stream>>>(W2, w2t, FF, D);
  k_router<<<NTOK / 8, 256, 0, stream>>>(x, Wr, br, meta);
  k_setup<<<1, 1, 0, stream>>>(meta, out);
  k_scatter<<<NTOK / 256, 256, 0, stream>>>(meta);

  size_t shmem = (size_t)(32 * SA + 32 * SH) * sizeof(u16) + MT * (sizeof(int) + sizeof(float));
  k_moe_ffn<<<MAXTILES, 256, shmem, stream>>>(w1t, w2t, xbuf, b1, b2, meta, out);
}
